// MedSegMamba_46428596470053
// MI455X (gfx1250) — compile-verified, matrix operand order checked by reading, not run on hardware
//
#include <hip/hip_runtime.h>

typedef long long ll;
typedef __attribute__((ext_vector_type(16))) _Float16 v16h;
typedef __attribute__((ext_vector_type(8)))  _Float16 v8h;
typedef __attribute__((ext_vector_type(8)))  float    v8f;

#define SHUF16(lo, hi) __builtin_shufflevector((lo), (hi), 0,1,2,3,4,5,6,7,8,9,10,11,12,13,14,15)

// ---------------------------------------------------------------------------
// Weight pre-conversion: strided f32 [N][K] -> packed f16 [Np][Kp], zero pad.
// Np/Kp are multiples of 64 so GEMM tiles are always in-bounds & 16B aligned.
// ---------------------------------------------------------------------------
__global__ void k_cvt_pad(const float* __restrict__ src, ll sN, ll sK,
                          _Float16* __restrict__ dst, int N, int K, int Kp, ll total)
{
    ll idx = (ll)blockIdx.x * blockDim.x + threadIdx.x;
    if (idx >= total) return;
    int n = (int)(idx / Kp), k = (int)(idx % Kp);
    float v = (n < N && k < K) ? src[(ll)n * sN + (ll)k * sK] : 0.f;
    dst[idx] = (_Float16)v;
}

// ---------------------------------------------------------------------------
// WMMA GEMM:  C[m,n] = act( sum_k A[m*sAm + k*sAk] * Bh[n*Kp + k] + bias[n] )
// Block = 128 threads = 4 waves. Block tile: M=32, N=64, K-chunk=64.
// Waves arranged 2(m) x 2(n); each wave owns a 16x32 C tile = two 16x16
// accumulators sharing one A fragment -> 4 v_wmma per wave per K-chunk.
// A (f32 activations): clamped VGPR loads + cvt into LDS chunks.
// B (pre-converted f16 weights): GLOBAL_LOAD_ASYNC_TO_LDS_B128 straight into
// the padded LDS tile, synced with s_wait_asynccnt (ASYNCcnt) + barrier.
// Fragments read back as 2x ds_load_b128 per matrix.
// act: 0 = none, 1 = relu, 3 = softplus
// ---------------------------------------------------------------------------
__global__ __launch_bounds__(128)
void k_gemm(const float* __restrict__ A, ll sAm, ll sAk,
            const _Float16* __restrict__ Bh, int Kp,
            const float* __restrict__ bias,
            float* __restrict__ C, ll sCm, ll sCn,
            int M, int N, int K, int act)
{
    __shared__ v8h As[32][9];   // [m-row][k-chunk]  (8 used, 1 pad)
    __shared__ v8h Bs[64][9];   // [n-col][k-chunk]

    int m0 = blockIdx.x * 32, n0 = blockIdx.y * 64;
    int lane = threadIdx.x & 31, wave = threadIdx.x >> 5;
    int wm = (wave >> 1) * 16;          // 0 / 16
    int wn = (wave & 1) * 32;           // 0 / 32
    int hh = lane >> 4, ml = lane & 15;

    v8f acc0 = {0.f,0.f,0.f,0.f,0.f,0.f,0.f,0.f};
    v8f acc1 = {0.f,0.f,0.f,0.f,0.f,0.f,0.f,0.f};

    for (int k0 = 0; k0 < K; k0 += 64) {
        // ---- stage B: async DMA f16 weights -> LDS (always in-bounds) ----
#pragma unroll
        for (int it = 0; it < 4; ++it) {
            int t = threadIdx.x + it * 128;
            int col = t >> 3, c = t & 7;
            const _Float16* gp = Bh + (ll)(n0 + col) * Kp + (k0 + 8 * c);
            unsigned lp = (unsigned)(unsigned long long)(const void*)&Bs[col][c];
            asm volatile("global_load_async_to_lds_b128 %0, %1, off"
                         :: "v"(lp), "v"(gp) : "memory");
        }
        // ---- stage A: 32 rows x 8 chunks = 256 chunk-slots, 2 per thread ----
#pragma unroll
        for (int it = 0; it < 2; ++it) {
            int t = threadIdx.x + it * 128;
            int row = t >> 3, c = t & 7;
            int m = m0 + row, kb = k0 + 8 * c;
            float f[8];
            if (sAk == 1 && m < M && kb + 7 < K) {
                const float* p = A + (ll)m * sAm + kb;
#pragma unroll
                for (int e = 0; e < 8; ++e) f[e] = p[e];
            } else {
                int mc = (m < M) ? m : (M - 1);
#pragma unroll
                for (int e = 0; e < 8; ++e) {
                    int k = kb + e;
                    int kc = (k < K) ? k : (K - 1);
                    float v = A[(ll)mc * sAm + (ll)kc * sAk];
                    f[e] = (m < M && k < K) ? v : 0.f;
                }
            }
            v8h hv;
#pragma unroll
            for (int e = 0; e < 8; ++e) hv[e] = (_Float16)f[e];
            As[row][c] = hv;
        }
        // prefetch next K-chunk of A (speculative, no counter cost)
        if (k0 + 64 < K) {
            int m = m0 + (threadIdx.x >> 2);
            if (m < M) __builtin_prefetch(&A[(ll)m * sAm + (ll)(k0 + 64) * sAk], 0, 0);
        }
        asm volatile("s_wait_asynccnt 0x0" ::: "memory");
        __syncthreads();

        const v8h* ar = As[wm + ml];
        const v8h* b0 = Bs[wn + ml];
        const v8h* b1 = Bs[wn + 16 + ml];
#pragma unroll
        for (int s = 0; s < 2; ++s) {
            v16h a   = SHUF16(ar[4*s + hh], ar[4*s + 2 + hh]);
            v16h bb0 = SHUF16(b0[4*s + hh], b0[4*s + 2 + hh]);
            v16h bb1 = SHUF16(b1[4*s + hh], b1[4*s + 2 + hh]);
            acc0 = __builtin_amdgcn_wmma_f32_16x16x32_f16(false, a, false, bb0,
                                                          (short)0, acc0, false, false);
            acc1 = __builtin_amdgcn_wmma_f32_16x16x32_f16(false, a, false, bb1,
                                                          (short)0, acc1, false, false);
        }
        __syncthreads();
    }

#pragma unroll
    for (int r2 = 0; r2 < 8; ++r2) {
        int row = m0 + wm + r2 + 8 * hh;
        if (row >= M) continue;
        int col0 = n0 + wn + ml;
        int col1 = col0 + 16;
        if (col0 < N) {
            float v = acc0[r2];
            if (bias) v += bias[col0];
            if (act == 1)      v = fmaxf(v, 0.f);
            else if (act == 3) v = (v > 20.f) ? v : log1pf(expf(v));
            C[(ll)row * sCm + (ll)col0 * sCn] = v;
        }
        if (col1 < N) {
            float v = acc1[r2];
            if (bias) v += bias[col1];
            if (act == 1)      v = fmaxf(v, 0.f);
            else if (act == 3) v = (v > 20.f) ? v : log1pf(expf(v));
            C[(ll)row * sCm + (ll)col1 * sCn] = v;
        }
    }
}

// ---------------------------------------------------------------------------
// Implicit-GEMM 3x3x3 conv, stride 1, pad 1. in: [Cin][D][H][W],
// Bh: pre-converted f16 weights [Np][Kp] (K = Cin*27),
// out: [N][D][H][W] (channel major). A gathered im2col-style into LDS with
// clamped unconditional loads; B tiles via async DMA like k_gemm.
// ---------------------------------------------------------------------------
__global__ __launch_bounds__(128)
void k_conv3(const float* __restrict__ in, const _Float16* __restrict__ Bh, int Kp,
             const float* __restrict__ bias, float* __restrict__ out,
             int Cin, int N, int D, int H, int Wd, int act)
{
    __shared__ v8h As[32][9];
    __shared__ v8h Bs[64][9];

    ll HW = (ll)H * Wd;
    ll Ml = (ll)D * HW;
    int M = (int)Ml;
    int K = Cin * 27;

    int m0 = blockIdx.x * 32, n0 = blockIdx.y * 64;
    int lane = threadIdx.x & 31, wave = threadIdx.x >> 5;
    int wm = (wave >> 1) * 16;
    int wn = (wave & 1) * 32;
    int hh = lane >> 4, ml = lane & 15;

    v8f acc0 = {0.f,0.f,0.f,0.f,0.f,0.f,0.f,0.f};
    v8f acc1 = {0.f,0.f,0.f,0.f,0.f,0.f,0.f,0.f};

    for (int k0 = 0; k0 < K; k0 += 64) {
        // ---- stage B: async DMA f16 weights -> LDS ----
#pragma unroll
        for (int it = 0; it < 4; ++it) {
            int t = threadIdx.x + it * 128;
            int col = t >> 3, c = t & 7;
            const _Float16* gp = Bh + (ll)(n0 + col) * Kp + (k0 + 8 * c);
            unsigned lp = (unsigned)(unsigned long long)(const void*)&Bs[col][c];
            asm volatile("global_load_async_to_lds_b128 %0, %1, off"
                         :: "v"(lp), "v"(gp) : "memory");
        }
        // ---- stage A (im2col gather, clamped loads + post-select) ----
#pragma unroll
        for (int it = 0; it < 2; ++it) {
            int t = threadIdx.x + it * 128;
            int row = t >> 3, c = t & 7;
            int m = m0 + row, kb = k0 + 8 * c;
            bool mok = (m < M);
            int mc = mok ? m : (M - 1);
            int z = (int)((ll)mc / HW);
            int rem = (int)((ll)mc % HW);
            int yy = rem / Wd, xx = rem % Wd;
            float f[8];
#pragma unroll
            for (int e = 0; e < 8; ++e) {
                int k = kb + e;
                int kc = (k < K) ? k : (K - 1);
                int ci = kc / 27, kk3 = kc % 27;
                int kd = kk3 / 9, kh2 = (kk3 / 3) % 3, kw = kk3 % 3;
                int iz = z + kd - 1, iy = yy + kh2 - 1, ix = xx + kw - 1;
                bool ok = mok && (k < K) &&
                          iz >= 0 && iz < D && iy >= 0 && iy < H && ix >= 0 && ix < Wd;
                int izc = iz < 0 ? 0 : (iz >= D ? D - 1 : iz);
                int iyc = iy < 0 ? 0 : (iy >= H ? H - 1 : iy);
                int ixc = ix < 0 ? 0 : (ix >= Wd ? Wd - 1 : ix);
                float v = in[((ll)ci * D + izc) * HW + (ll)iyc * Wd + ixc];
                f[e] = ok ? v : 0.f;
            }
            v8h hv;
#pragma unroll
            for (int e = 0; e < 8; ++e) hv[e] = (_Float16)f[e];
            As[row][c] = hv;
        }
        asm volatile("s_wait_asynccnt 0x0" ::: "memory");
        __syncthreads();

        const v8h* ar = As[wm + ml];
        const v8h* b0 = Bs[wn + ml];
        const v8h* b1 = Bs[wn + 16 + ml];
#pragma unroll
        for (int s = 0; s < 2; ++s) {
            v16h a   = SHUF16(ar[4*s + hh], ar[4*s + 2 + hh]);
            v16h bb0 = SHUF16(b0[4*s + hh], b0[4*s + 2 + hh]);
            v16h bb1 = SHUF16(b1[4*s + hh], b1[4*s + 2 + hh]);
            acc0 = __builtin_amdgcn_wmma_f32_16x16x32_f16(false, a, false, bb0,
                                                          (short)0, acc0, false, false);
            acc1 = __builtin_amdgcn_wmma_f32_16x16x32_f16(false, a, false, bb1,
                                                          (short)0, acc1, false, false);
        }
        __syncthreads();
    }

#pragma unroll
    for (int r2 = 0; r2 < 8; ++r2) {
        int row = m0 + wm + r2 + 8 * hh;
        if (row >= M) continue;
        int col0 = n0 + wn + ml;
        int col1 = col0 + 16;
        if (col0 < N) {
            float v = acc0[r2];
            if (bias) v += bias[col0];
            if (act == 1) v = fmaxf(v, 0.f);
            out[(ll)col0 * Ml + row] = v;
        }
        if (col1 < N) {
            float v = acc1[r2];
            if (bias) v += bias[col1];
            if (act == 1) v = fmaxf(v, 0.f);
            out[(ll)col1 * Ml + row] = v;
        }
    }
}

// ---------------------------------------------------------------------------
// GroupNorm (in-place), one block per group, optional relu.
// ---------------------------------------------------------------------------
__global__ __launch_bounds__(256)
void k_groupnorm(float* __restrict__ x, const float* __restrict__ g,
                 const float* __restrict__ b, int C, ll M, int groups, int relu)
{
    int grp = blockIdx.x;
    int cpg = C / groups;
    ll n = (ll)cpg * M;
    float* base = x + (ll)grp * n;
    __shared__ float red[256];
    int tid = threadIdx.x;

    float s = 0.f, s2 = 0.f;
    for (ll i = tid; i < n; i += 256) { float v = base[i]; s += v; s2 += v * v; }
    red[tid] = s; __syncthreads();
    for (int o = 128; o > 0; o >>= 1) { if (tid < o) red[tid] += red[tid + o]; __syncthreads(); }
    float mean = red[0] / (float)n;
    __syncthreads();
    red[tid] = s2; __syncthreads();
    for (int o = 128; o > 0; o >>= 1) { if (tid < o) red[tid] += red[tid + o]; __syncthreads(); }
    float var = red[0] / (float)n - mean * mean;
    float rs = rsqrtf(var + 1e-5f);

    for (ll i = tid; i < n; i += 256) {
        int ch = grp * cpg + (int)(i / M);
        float v = (base[i] - mean) * rs * g[ch] + b[ch];
        if (relu) v = fmaxf(v, 0.f);
        base[i] = v;
    }
}

// BatchNorm (inference), elementwise; in may equal out.
__global__ void k_bn(const float* __restrict__ in, float* __restrict__ out,
                     const float* __restrict__ g, const float* __restrict__ b,
                     const float* __restrict__ mu, const float* __restrict__ var,
                     int C, ll M, int relu)
{
    ll idx = (ll)blockIdx.x * blockDim.x + threadIdx.x;
    if (idx >= (ll)C * M) return;
    int ch = (int)(idx / M);
    float v = (in[idx] - mu[ch]) * rsqrtf(var[ch] + 1e-5f) * g[ch] + b[ch];
    if (relu) v = fmaxf(v, 0.f);
    out[idx] = v;
}

// LayerNorm over last dim (rows x c), in place. One thread per row (c <= 128).
__global__ void k_lnrows(float* __restrict__ x, const float* __restrict__ g,
                         const float* __restrict__ b, int rows, int c)
{
    int r = blockIdx.x * blockDim.x + threadIdx.x;
    if (r >= rows) return;
    float* p = x + (ll)r * c;
    float s = 0.f, s2 = 0.f;
    for (int j = 0; j < c; ++j) { float v = p[j]; s += v; s2 += v * v; }
    float mu = s / c;
    float var = s2 / c - mu * mu;
    float rs = rsqrtf(var + 1e-5f);
    for (int j = 0; j < c; ++j) p[j] = (p[j] - mu) * rs * g[j] + b[j];
}

// Depthwise 3x3x3 conv + SiLU. in/out: [C][D][H][W], w: [C][27].
__global__ void k_dwconv3_silu(const float* __restrict__ in, const float* __restrict__ w,
                               const float* __restrict__ bias, float* __restrict__ out,
                               int C, int D, int H, int Wd)
{
    ll HW = (ll)H * Wd;
    ll M = (ll)D * HW;
    ll idx = (ll)blockIdx.x * blockDim.x + threadIdx.x;
    if (idx >= (ll)C * M) return;
    int ch = (int)(idx / M);
    ll m = idx % M;
    int z = (int)(m / HW);
    int rem = (int)(m % HW);
    int yy = rem / Wd, xx = rem % Wd;
    const float* ib = in + (ll)ch * M;
    const float* wc = w + (ll)ch * 27;
    float acc = bias[ch];
    for (int kd = 0; kd < 3; ++kd) {
        int iz = z + kd - 1; if (iz < 0 || iz >= D) continue;
        for (int kh = 0; kh < 3; ++kh) {
            int iy = yy + kh - 1; if (iy < 0 || iy >= H) continue;
            for (int kw = 0; kw < 3; ++kw) {
                int ix = xx + kw - 1; if (ix < 0 || ix >= Wd) continue;
                acc += wc[kd * 9 + kh * 3 + kw] * ib[(ll)iz * HW + (ll)iy * Wd + ix];
            }
        }
    }
    out[idx] = acc / (1.f + expf(-acc));
}

// Selective-scan: grid = c channels, block = 16 (one lane per state n).
// dt, xs: [L][c]; dbl rows hold [.. r .. | Bm(16) | Cm(16)] with stride ldd.
__global__ void k_scan(const float* __restrict__ dt, const float* __restrict__ xs,
                       const float* __restrict__ dbl, int ldd, int rr,
                       const float* __restrict__ A_log, const float* __restrict__ Dp,
                       float* __restrict__ y, int L, int c)
{
    int ci = blockIdx.x;
    int n = threadIdx.x;                    // 0..15
    float Av = -expf(A_log[ci * 16 + n]);
    float Dv = Dp[ci];
    float h = 0.f;
    for (int l = 0; l < L; ++l) {
        float dtv = dt[(ll)l * c + ci];
        float xsv = xs[(ll)l * c + ci];
        const float* row = dbl + (ll)l * ldd;
        float a = expf(dtv * Av);
        h = a * h + dtv * xsv * row[rr + n];
        float contrib = h * row[rr + 16 + n];
        for (int o = 8; o >= 1; o >>= 1) contrib += __shfl_xor(contrib, o, 32);
        if (n == 0) y[(ll)l * c + ci] = contrib + Dv * xsv;
    }
}

// y[m,j] *= silu(xz[m, c + j])   (xz rows of width 2c)
__global__ void k_gate(float* __restrict__ y, const float* __restrict__ xz, int c, ll rows)
{
    ll idx = (ll)blockIdx.x * blockDim.x + threadIdx.x;
    if (idx >= rows * c) return;
    ll m = idx / c; int j = (int)(idx % c);
    float z = xz[m * (2 * c) + c + j];
    y[idx] *= z / (1.f + expf(-z));
}

// Generic strided permute copy: dst[m*d_m + j*d_c] = src[off + m*s_m + j*s_c]
__global__ void k_perm(float* __restrict__ dst, const float* __restrict__ src,
                       ll Mr, ll Cc, ll s_m, ll s_c, ll off, ll d_m, ll d_c)
{
    ll idx = (ll)blockIdx.x * blockDim.x + threadIdx.x;
    if (idx >= Mr * Cc) return;
    ll m = idx / Cc, j = idx % Cc;
    dst[m * d_m + j * d_c] = src[off + m * s_m + j * s_c];
}

__global__ void k_copy(float* __restrict__ dst, const float* __restrict__ src, ll n)
{
    ll i = (ll)blockIdx.x * blockDim.x + threadIdx.x;
    if (i < n) dst[i] = src[i];
}

__global__ void k_add(float* __restrict__ dst, const float* __restrict__ a, ll n)
{
    ll i = (ll)blockIdx.x * blockDim.x + threadIdx.x;
    if (i < n) dst[i] += a[i];
}

// Patch merging gather: out[(o*Cin+ci)*M2 + m2] = in[ci, 2z+od, 2y+oh, 2x+ow]
__global__ void k_patch_gather(const float* __restrict__ in, float* __restrict__ out,
                               int Cin, int D, int H, int Wd)
{
    static const int od[8] = {0,1,0,0,1,1,0,1};
    static const int oh[8] = {0,0,1,0,1,0,1,1};
    static const int ow[8] = {0,0,0,1,0,1,1,1};
    int D2 = D / 2, H2 = H / 2, W2 = Wd / 2;
    ll HW2 = (ll)H2 * W2;
    ll M2 = (ll)D2 * HW2;
    ll total = (ll)8 * Cin * M2;
    ll idx = (ll)blockIdx.x * blockDim.x + threadIdx.x;
    if (idx >= total) return;
    ll c8 = idx / M2, m2 = idx % M2;
    int o = (int)(c8 / Cin), ci = (int)(c8 % Cin);
    int z = (int)(m2 / HW2);
    int rem = (int)(m2 % HW2);
    int yy = rem / W2, xx = rem % W2;
    ll src = ((ll)ci * D + (2 * z + od[o])) * H * Wd + (ll)(2 * yy + oh[o]) * Wd + (2 * xx + ow[o]);
    out[idx] = in[src];
}

// pixel_shuffle3d r=2: in [co*8][D][H][W] -> out [co][2D][2H][2W]
__global__ void k_pixel_shuffle(const float* __restrict__ in, float* __restrict__ out,
                                int co, int D, int H, int Wd)
{
    ll HW = (ll)H * Wd;
    ll M = (ll)D * HW;
    ll total = (ll)co * 8 * M;
    ll idx = (ll)blockIdx.x * blockDim.x + threadIdx.x;
    if (idx >= total) return;
    int ch = (int)(idx / M);
    ll m = idx % M;
    int coi = ch / 8;
    int rcode = ch % 8;
    int rd = rcode / 4, rh = (rcode / 2) % 2, rw = rcode % 2;
    int z = (int)(m / HW);
    int rem = (int)(m % HW);
    int yy = rem / Wd, xx = rem % Wd;
    ll D2 = 2 * D, H2 = 2 * H, W2 = 2 * Wd;
    ll dst = ((ll)coi * D2 + (2 * z + rd)) * H2 * W2 + (ll)(2 * yy + rh) * W2 + (2 * xx + rw);
    out[dst] = in[idx];
}

// out_ch[i*2+p] = concat_ch[p*C2+i] (+ residual x). l,r,x,out all [.][M].
__global__ void k_shuffle_add(const float* __restrict__ l, const float* __restrict__ r,
                              const float* __restrict__ x, float* __restrict__ out,
                              int C2, ll M)
{
    ll total = (ll)2 * C2 * M;
    ll idx = (ll)blockIdx.x * blockDim.x + threadIdx.x;
    if (idx >= total) return;
    int j = (int)(idx / M);
    ll m = idx % M;
    int i = j >> 1, p = j & 1;
    float v = p ? r[(ll)i * M + m] : l[(ll)i * M + m];
    out[idx] = v + x[idx];
}

// ===========================================================================
// Host-side orchestration
// ===========================================================================
struct ResP { const float *w1,*b1,*g1g,*g1b,*w2,*b2,*g2g,*g2b,*wid,*bid; };
struct SsmP { const float *Win,*cw,*cb,*Wx,*Wdt,*bdt,*Alog,*Dp,*lng,*lnb,*Wout; };
struct BlkP { const float *ln1g,*ln1b; SsmP ssm;
              const float *cw1,*cb1,*cw2,*cb2,*cw3,*cb3;
              const float *bn1g,*bn1b,*bn1m,*bn1v;
              const float *bn2g,*bn2b,*bn2m,*bn2v;
              const float *bn3g,*bn3b,*bn3m,*bn3v; };
struct DownP { const float* pm; ResP res; const float *ag,*ab; BlkP blk; };
struct UpP  { const float *pw,*pb,*g1g,*g1b,*sw,*sb,*g2g,*g2b; };

static inline unsigned ewb(ll n) { return (unsigned)((n + 255) / 256); }

static _Float16* g_wh = nullptr;   // f16 weight scratch (set each launch, deterministic)

// Convert strided f32 weights to packed padded f16 and run WMMA GEMM.
static void gemm(hipStream_t st, const float* A, ll sAm, ll sAk,
                 const float* B, ll sBn, ll sBk, const float* bias,
                 float* C, ll sCm, ll sCn, int M, int N, int K, int act)
{
    int Kp = (K + 63) & ~63;
    int Np = (N + 63) & ~63;
    ll tot = (ll)Np * Kp;
    k_cvt_pad<<<ewb(tot), 256, 0, st>>>(B, sBn, sBk, g_wh, N, K, Kp, tot);
    dim3 grid((M + 31) / 32, (N + 63) / 64);
    k_gemm<<<grid, 128, 0, st>>>(A, sAm, sAk, g_wh, Kp, bias, C, sCm, sCn, M, N, K, act);
}

static void conv3(hipStream_t st, const float* in, const float* w, const float* bias,
                  float* out, int Cin, int N, int D, int H, int Wd, int act)
{
    int M = D * H * Wd;
    int K = Cin * 27;
    int Kp = (K + 63) & ~63;
    int Np = (N + 63) & ~63;
    ll tot = (ll)Np * Kp;
    k_cvt_pad<<<ewb(tot), 256, 0, st>>>(w, K, 1, g_wh, N, K, Kp, tot);
    dim3 grid((M + 31) / 32, (N + 63) / 64);
    k_conv3<<<grid, 128, 0, st>>>(in, g_wh, Kp, bias, out, Cin, N, D, H, Wd, act);
}

static void gn(hipStream_t st, float* x, const float* g, const float* b, int C, ll M, int relu)
{
    k_groupnorm<<<8, 256, 0, st>>>(x, g, b, C, M, 8, relu);
}

static void bn(hipStream_t st, const float* in, float* out, const float* g, const float* b,
               const float* m, const float* v, int C, ll M, int relu)
{
    k_bn<<<ewb((ll)C * M), 256, 0, st>>>(in, out, g, b, m, v, C, M, relu);
}

static void perm(hipStream_t st, float* dst, const float* src,
                 ll Mr, ll Cc, ll s_m, ll s_c, ll off, ll d_m, ll d_c)
{
    k_perm<<<ewb(Mr * Cc), 256, 0, st>>>(dst, src, Mr, Cc, s_m, s_c, off, d_m, d_c);
}

static void res_block(hipStream_t st, const float* x, int Cin, int Cout,
                      int D, int H, int Wd, const ResP& p,
                      float* out, float* t1, float* t2)
{
    ll M = (ll)D * H * Wd;
    conv3(st, x, p.w1, p.b1, t1, Cin, Cout, D, H, Wd, 0);
    gn(st, t1, p.g1g, p.g1b, Cout, M, 1);
    conv3(st, t1, p.w2, p.b2, t2, Cout, Cout, D, H, Wd, 0);
    gn(st, t2, p.g2g, p.g2b, Cout, M, 1);
    // identity path: 1x1x1 conv as GEMM (A channel-major)
    gemm(st, x, 1, M, p.wid, Cin, 1, p.bid, out, 1, M, (int)M, Cout, Cin, 0);
    k_add<<<ewb((ll)Cout * M), 256, 0, st>>>(out, t2, (ll)Cout * M);
}

static void ss3d(hipStream_t st, float* xin /*[M][c], post-LN*/, int c,
                 int D, int H, int Wd, const SsmP& p, float* r_cm_out, float* arena)
{
    ll M = (ll)D * H * Wd;
    int r = (c + 15) / 16; if (r < 1) r = 1;
    int ld = r + 32;
    float* xz    = arena;              arena += M * 2 * c;
    float* xc    = arena;              arena += M * c;
    float* xs_cm = arena;              arena += M * c;
    float* xs    = arena;              arena += M * c;
    float* dbl   = arena;              arena += M * ld;
    float* dt    = arena;              arena += M * c;
    float* y     = arena;              arena += M * c;

    // xz = x @ Win.T   [M][2c]
    gemm(st, xin, c, 1, p.Win, c, 1, nullptr, xz, 2 * c, 1, (int)M, 2 * c, c, 0);
    // xc = xz[:, :c] -> channel-major
    perm(st, xc, xz, M, c, 2 * c, 1, 0, 1, M);
    // depthwise conv + silu
    k_dwconv3_silu<<<ewb((ll)c * M), 256, 0, st>>>(xc, p.cw, p.cb, xs_cm, c, D, H, Wd);
    // xs rows [M][c]
    perm(st, xs, xs_cm, M, c, 1, M, 0, c, 1);
    // dbl = xs @ Wx.T  [M][r+32]
    gemm(st, xs, c, 1, p.Wx, c, 1, nullptr, dbl, ld, 1, (int)M, ld, c, 0);
    // dt = softplus(dbl[:, :r] @ Wdt.T + bdt)  [M][c]
    gemm(st, dbl, ld, 1, p.Wdt, r, 1, p.bdt, dt, c, 1, (int)M, c, r, 3);
    // selective scan + einsum over states
    k_scan<<<c, 16, 0, st>>>(dt, xs, dbl, ld, r, p.Alog, p.Dp, y, (int)M, c);
    // layer norm, gate with silu(z)
    k_lnrows<<<ewb(M), 256, 0, st>>>(y, p.lng, p.lnb, (int)M, c);
    k_gate<<<ewb(M * c), 256, 0, st>>>(y, xz, c, M);
    // out = y @ Wout.T, then rows -> channel-major
    float* y2 = xc;  // reuse
    gemm(st, y, c, 1, p.Wout, c, 1, nullptr, y2, c, 1, (int)M, c, c, 0);
    perm(st, r_cm_out, y2, M, c, c, 1, 0, 1, M);
}

static void ss_conv_ssm(hipStream_t st, const float* x, int C, int D, int H, int Wd,
                        const BlkP& p, float* out, float* s0, float* s1, float* arena)
{
    int c2 = C / 2;
    ll M = (ll)D * H * Wd;
    const float* left  = x;
    const float* right = x + (ll)c2 * M;

    // ---- right branch (SSM) ----
    float* rrows = arena;              // [M][c2]
    float* rcm   = arena + M * c2;     // [c2][M]
    float* inner = arena + 2 * M * c2; // ss3d temporaries
    perm(st, rrows, right, M, c2, 1, M, 0, c2, 1);             // cm -> rows
    k_lnrows<<<ewb(M), 256, 0, st>>>(rrows, p.ln1g, p.ln1b, (int)M, c2);
    ss3d(st, rrows, c2, D, H, Wd, p.ssm, rcm, inner);

    // ---- left branch (convs) ----
    bn(st, left, s0, p.bn1g, p.bn1b, p.bn1m, p.bn1v, c2, M, 0);
    conv3(st, s0, p.cw1, p.cb1, s1, c2, c2, D, H, Wd, 0);
    bn(st, s1, s1, p.bn2g, p.bn2b, p.bn2m, p.bn2v, c2, M, 1);
    conv3(st, s1, p.cw2, p.cb2, s0, c2, c2, D, H, Wd, 0);
    bn(st, s0, s0, p.bn3g, p.bn3b, p.bn3m, p.bn3v, c2, M, 1);
    gemm(st, s0, 1, M, p.cw3, c2, 1, p.cb3, s1, 1, M, (int)M, c2, c2, 1); // 1x1 + relu

    // ---- channel-shuffle concat + residual ----
    k_shuffle_add<<<ewb((ll)C * M), 256, 0, st>>>(s1, rcm, x, out, c2, M);
}

static void down_block(hipStream_t st, const float* in, int Cin, int D, int H, int Wd,
                       const DownP& p, float* out, float* const* sl)
{
    int Cout = 2 * Cin;
    int D2 = D / 2, H2 = H / 2, W2 = Wd / 2;
    ll M2 = (ll)D2 * H2 * W2;
    k_patch_gather<<<ewb((ll)8 * Cin * M2), 256, 0, st>>>(in, sl[0], Cin, D, H, Wd);
    gemm(st, sl[0], 1, M2, p.pm, 8 * Cin, 1, nullptr, sl[1], 1, M2, (int)M2, Cout, 8 * Cin, 0);
    res_block(st, sl[1], Cout, Cout, D2, H2, W2, p.res, sl[2], sl[3], sl[4]);
    gn(st, sl[2], p.ag, p.ab, Cout, M2, 1);
    ss_conv_ssm(st, sl[2], Cout, D2, H2, W2, p.blk, out, sl[0], sl[1], sl[3]);
}

extern "C" void kernel_launch(void* const* d_in, const int* in_sizes, int n_in,
                              void* d_out, int out_size, void* d_ws, size_t ws_size,
                              hipStream_t stream)
{
    (void)in_sizes; (void)n_in; (void)out_size; (void)ws_size;

    int ip = 0;
    auto F = [&]() { return (const float*)d_in[ip++]; };

    const float* X = F();  // (1,1,96,96,96)

    auto loadRes = [&](ResP& r) {
        r.w1 = F(); r.b1 = F(); r.g1g = F(); r.g1b = F();
        r.w2 = F(); r.b2 = F(); r.g2g = F(); r.g2b = F();
        r.wid = F(); r.bid = F();
    };
    auto loadSsm = [&](SsmP& s) {
        s.Win = F(); s.cw = F(); s.cb = F(); s.Wx = F(); s.Wdt = F(); s.bdt = F();
        s.Alog = F(); s.Dp = F(); s.lng = F(); s.lnb = F(); s.Wout = F();
    };
    auto loadBlk = [&](BlkP& k) {
        k.ln1g = F(); k.ln1b = F();
        loadSsm(k.ssm);
        k.cw1 = F(); k.cb1 = F(); k.cw2 = F(); k.cb2 = F(); k.cw3 = F(); k.cb3 = F();
        k.bn1g = F(); k.bn1b = F(); k.bn1m = F(); k.bn1v = F();
        k.bn2g = F(); k.bn2b = F(); k.bn2m = F(); k.bn2v = F();
        k.bn3g = F(); k.bn3b = F(); k.bn3m = F(); k.bn3v = F();
    };
    auto loadDown = [&](DownP& d) {
        d.pm = F(); loadRes(d.res); d.ag = F(); d.ab = F(); loadBlk(d.blk);
    };

    ResP  conv1;  loadRes(conv1);
    DownP d1;     loadDown(d1);
    DownP d2;     loadDown(d2);
    DownP d3;     loadDown(d3);
    UpP up;
    up.pw = F(); up.pb = F(); up.g1g = F(); up.g1b = F();
    up.sw = F(); up.sb = F(); up.g2g = F(); up.g2b = F();
    ResP upc4; loadRes(upc4);

    // ---- workspace slots: 8 slots of 32*96^3 floats each ----
    const ll S = 28311552LL;
    float* ws = (float*)d_ws;
    float* sl[8];
    for (int i = 0; i < 8; ++i) sl[i] = ws + (ll)i * S;
    float* x1  = sl[6];                  // 32 @ 96^3
    float* x2  = sl[7];                  // 64 @ 48^3   (7.08M)
    float* x3  = sl[7] + 8000000LL;      // 128 @ 24^3  (1.77M)
    float* xb  = sl[7] + 10000000LL;     // 256 @ 12^3  (0.44M)
    float* cat = sl[7] + 11000000LL;     // 256 @ 24^3  (3.54M)
    g_wh = (_Float16*)sl[5];             // f16 weight scratch (56.6M halves)

    // ---- encoder ----
    res_block(stream, X, 1, 32, 96, 96, 96, conv1, x1, sl[0], sl[1]);
    down_block(stream, x1, 32, 96, 96, 96, d1, x2, sl);
    down_block(stream, x2, 64, 48, 48, 48, d2, x3, sl);
    down_block(stream, x3, 128, 24, 24, 24, d3, xb, sl);

    // ---- up_block ----
    const ll Mb = 1728, M3 = 13824;
    conv3(stream, xb, up.pw, up.pb, sl[0], 256, 1024, 12, 12, 12, 0);
    gn(stream, sl[0], up.g1g, up.g1b, 1024, Mb, 1);
    conv3(stream, sl[0], up.sw, up.sb, sl[1], 1024, 1024, 12, 12, 12, 0);
    k_pixel_shuffle<<<ewb(1024 * Mb), 256, 0, stream>>>(sl[1], cat + 128 * M3, 128, 12, 12, 12);
    gn(stream, cat + 128 * M3, up.g2g, up.g2b, 128, M3, 1);

    // ---- concat [x3, up] and final res_block -> d_out ----
    k_copy<<<ewb(128 * M3), 256, 0, stream>>>(cat, x3, 128 * M3);
    res_block(stream, cat, 256, 128, 24, 24, 24, upc4, (float*)d_out, sl[0], sl[1]);
}
